// OriginalHypernetMultiheadAttention_86861418595096
// MI455X (gfx1250) — matrix operand order repaired and checked by reading, not compiled, not run on hardware
//
#include <hip/hip_runtime.h>
#include <hip/hip_bf16.h>

// ---------------------------------------------------------------------------
// CDNA5 (gfx1250) fused hypernet-attention.
//   B=2, L=1024, E=512, H=8, D=64  (hardcoded; shapes are static in reference)
// Pipeline:
//   [cvt x4]  W* f32 -> bf16 once (kills 32x redundant per-block conversion)
//   [gemm x3] q/k/v = in @ W^T + b      (bf16 WMMA, bf16 out to workspace)
//   [attn]    scores -> rms_head -> w -> value-mix -> silu -> mix  (fused)
//   [gemm]    out = attn @ Wo^T + bo    (bf16 WMMA, f32 out)
// bf16 tiles are staged to LDS with GLOBAL_LOAD_ASYNC_TO_LDS_B128 (ASYNCcnt)
// when the toolchain exposes the builtin; falls back to uint4 copies.
// ---------------------------------------------------------------------------

typedef __attribute__((ext_vector_type(16))) __bf16 v16bf;
typedef __attribute__((ext_vector_type(8)))  float  v8f;
typedef __attribute__((ext_vector_type(4)))  int    v4i;

#if defined(__gfx1250__) && __has_builtin(__builtin_amdgcn_global_load_async_to_lds_b128)
#define HAS_ASYNC_LDS 1
#else
#define HAS_ASYNC_LDS 0
#endif

__device__ __forceinline__ unsigned short f32_to_bf16(float f) {
    union { float f; unsigned int u; } c; c.f = f;
    unsigned int r = c.u + 0x7FFFu + ((c.u >> 16) & 1u);   // round-nearest-even
    return (unsigned short)(r >> 16);
}
__device__ __forceinline__ float bf16_to_f32(unsigned short h) {
    union { unsigned int u; float f; } c; c.u = ((unsigned int)h) << 16;
    return c.f;
}

// 16-byte global(bf16) -> LDS copy; async-to-LDS on CDNA5.
// Caller guarantees 16B alignment of both addresses.
__device__ __forceinline__ void copy16B(const unsigned short* __restrict__ g,
                                        unsigned short* l) {
#if HAS_ASYNC_LDS
    __builtin_amdgcn_global_load_async_to_lds_b128(
        (__attribute__((address_space(1))) v4i*)g,
        (__attribute__((address_space(3))) v4i*)l, 0, 0);
#else
    *(uint4*)l = *(const uint4*)g;
#endif
}
__device__ __forceinline__ void wait_async_lds() {
#if HAS_ASYNC_LDS
#if __has_builtin(__builtin_amdgcn_s_wait_asynccnt)
    __builtin_amdgcn_s_wait_asynccnt(0);
#else
    asm volatile("s_wait_asynccnt 0x0" ::: "memory");
#endif
#endif
}

// ---------------------------------------------------------------------------
// One-shot f32 -> bf16 conversion (for the four weight matrices).
// ---------------------------------------------------------------------------
__global__ __launch_bounds__(256) void cvt_f32_bf16_kernel(
    const float* __restrict__ src, unsigned short* __restrict__ dst, int n)
{
    int i = (blockIdx.x * 256 + threadIdx.x) * 4;
    if (i + 3 < n) {
        float4 f = *(const float4*)(src + i);
        uint2 o;
        o.x = (unsigned int)f32_to_bf16(f.x) | ((unsigned int)f32_to_bf16(f.y) << 16);
        o.y = (unsigned int)f32_to_bf16(f.z) | ((unsigned int)f32_to_bf16(f.w) << 16);
        *(uint2*)(dst + i) = o;
    }
}

// ---------------------------------------------------------------------------
// C[m][n] = sum_k A[m][k] * W[n][k] + bias[n]
// A: MxK row-major (float, or raw-bf16 ushort), Wb: NxK row-major bf16 bits.
// Block tile 64x128, 8 waves (4x2), each wave 16x64 = 4 WMMA accumulators.
// LDA = 40 bf16 (80B rows): 16B-aligned chunks for async b128 LDS writes,
// 20-dword pitch spreads 16 consecutive rows across distinct banks.
// ---------------------------------------------------------------------------
template <typename AT, bool OUT_BF16>
__global__ __launch_bounds__(256) void gemm_bt_kernel(
    const AT* __restrict__ A, const unsigned short* __restrict__ Wb,
    const float* __restrict__ bias, void* __restrict__ Cout,
    int M, int N, int K)
{
    constexpr int BM = 64, BN = 128, BK = 32, LDA = 40;
    constexpr bool A_BF16 = (sizeof(AT) == 2);
    __shared__ unsigned short As[BM * LDA];
    __shared__ unsigned short Bs[BN * LDA];

    const int tid  = threadIdx.x;
    const int wave = tid >> 5, lane = tid & 31;
    const int wm   = wave & 3, wn = wave >> 2;           // 4(M) x 2(N) waves
    const int m0   = blockIdx.x * BM, n0 = blockIdx.y * BN;

    v8f acc[4] = {};                                     // 16 x 64 per wave

    const int ra = tid >> 2, ca = (tid & 3) * 8;         // A: 8 elems/thread
    const int rb = tid >> 1, cb = (tid & 1) * 16;        // B: 16 elems/thread

    for (int kk = 0; kk < K; kk += BK) {
        // ---- stage A tile 64x32
        if constexpr (A_BF16) {
            copy16B((const unsigned short*)A + (size_t)(m0 + ra) * K + kk + ca,
                    &As[ra * LDA + ca]);
        } else {
            const float* src = (const float*)A + (size_t)(m0 + ra) * K + kk + ca;
            unsigned short* dst = &As[ra * LDA + ca];
#pragma unroll
            for (int j = 0; j < 8; ++j) dst[j] = f32_to_bf16(src[j]);
        }
        // ---- stage B tile 128x32 (bf16 weights: pure byte copy)
        {
            const unsigned short* src = Wb + (size_t)(n0 + rb) * K + kk + cb;
            unsigned short* dst = &Bs[rb * LDA + cb];
            copy16B(src, dst);
            copy16B(src + 8, dst + 8);
        }
        if (kk + BK < K) {                               // L2 prefetch next slab
            __builtin_prefetch((const AT*)A + (size_t)(m0 + ra) * K + kk + BK, 0, 1);
            __builtin_prefetch(Wb + (size_t)(n0 + rb) * K + kk + BK, 0, 1);
        }
        wait_async_lds();
        __syncthreads();

        // ---- pack A fragment (16x32, contiguous k per lane half):
        //   lane<16:  row=lane,    K = 0..15
        //   lane>=16: row=lane-16, K = 16..31
        union { v16bf v; unsigned int u[8]; } afrag;
        {
            int row = wm * 16 + (lane & 15);
            int kb  = (lane < 16) ? 0 : 16;
            const unsigned int* p = (const unsigned int*)&As[row * LDA];
#pragma unroll
            for (int i = 0; i < 8; ++i) {
                afrag.u[i]     = p[(kb >> 1) + i];
            }
        }
        // ---- 4 N-subtiles: pack B (32x16: lane holds col N=lane%16,
        //      K = lane<16 ? 0..15 : 16..31) and accumulate
#pragma unroll
        for (int t = 0; t < 4; ++t) {
            union { v16bf v; unsigned int u[8]; } bfrag;
            int col = wn * 64 + t * 16 + (lane & 15);
            int kb2 = (lane < 16) ? 0 : 16;
            const unsigned int* p = (const unsigned int*)&Bs[col * LDA + kb2];
#pragma unroll
            for (int i = 0; i < 8; ++i) bfrag.u[i] = p[i];
            acc[t] = __builtin_amdgcn_wmma_f32_16x16x32_bf16(
                false, afrag.v, false, bfrag.v, (short)0, acc[t], false, false);
        }
        __syncthreads();
    }

    // ---- epilogue: C layout VGPR r -> M = r + 8*(lane>=16), N = lane%16
    const int mbase = m0 + wm * 16 + ((lane < 16) ? 0 : 8);
    const int nbase = n0 + wn * 64 + (lane & 15);
#pragma unroll
    for (int t = 0; t < 4; ++t) {
        int n = nbase + t * 16;
        float bv = bias[n];
#pragma unroll
        for (int r = 0; r < 8; ++r) {
            float val = acc[t][r] + bv;
            size_t idx = (size_t)(mbase + r) * N + n;
            if (OUT_BF16) ((unsigned short*)Cout)[idx] = f32_to_bf16(val);
            else          ((float*)Cout)[idx]          = val;
        }
    }
}

// ---------------------------------------------------------------------------
// Fused attention middle. One workgroup per (batch, 16-query tile).
// 8 waves = 8 heads for the WMMA score phase; all 256 lanes for the mix phase.
// ---------------------------------------------------------------------------
__global__ __launch_bounds__(256) void attn_kernel(
    const unsigned short* __restrict__ Q,   // [B*L, 512] bf16 bits
    const unsigned short* __restrict__ Kb,
    const unsigned short* __restrict__ Vb,
    unsigned short* __restrict__ Obf)       // [B*L, 512] bf16 bits
{
    constexpr int E = 512, LDT = 520;       // 1040B rows: 16B aligned + bank spread
    __shared__ unsigned short q_s[16 * LDT];
    __shared__ unsigned short k_s[16 * LDT];
    __shared__ unsigned short v_s[16 * LDT];
    __shared__ float w_s[8 * 256];          // w[h][q][k] for current tile
    __shared__ float ssq[256];              // sum_h s^2 at each (q,k)

    const int tid  = threadIdx.x;
    const int wave = tid >> 5, lane = tid & 31;
    const int h    = wave;                  // head owned by this wave
    const int b    = blockIdx.x >> 6;
    const int qt   = blockIdx.x & 63;
    const size_t qrow0 = (size_t)(b * 1024 + qt * 16) * E;
    const size_t krow0 = (size_t)(b * 1024) * E;

    const int r_ld = tid >> 4, c_ld = (tid & 15) * 32;   // tile-stage mapping

    // ---- load query tile (16x512) via async-to-LDS
    {
        const unsigned short* src = Q + qrow0 + (size_t)r_ld * E + c_ld;
        unsigned short* dst = &q_s[r_ld * LDT + c_ld];
#pragma unroll
        for (int j = 0; j < 4; ++j) copy16B(src + j * 8, dst + j * 8);
    }
    wait_async_lds();
    __syncthreads();

    // ---- pack this head's A fragments once (d 0..31 and d 32..63)
    union { v16bf v; unsigned int u[8]; } a0, a1;
    {
        int row = lane & 15;
        int kb  = (lane < 16) ? 0 : 16;
        const unsigned int* p = (const unsigned int*)&q_s[row * LDT + h * 64];
#pragma unroll
        for (int i = 0; i < 8; ++i) {
            a0.u[i]     = p[(kb >> 1) + i];
            a1.u[i]     = p[16 + (kb >> 1) + i];
        }
    }

    const int q_my = tid >> 4;              // mix phase: each lane owns (q, 4 d)
    const int d_my = (tid & 15) * 4;
    float oacc[8][4] = {};                  // out[q_my][h][d_my..+3]

    for (int kt = 0; kt < 64; ++kt) {
        // ---- stage key/value tiles (16x512 each, async) + reset ssq
        {
            const unsigned short* ks = Kb + krow0 + (size_t)(kt * 16 + r_ld) * E + c_ld;
            const unsigned short* vs = Vb + krow0 + (size_t)(kt * 16 + r_ld) * E + c_ld;
            unsigned short* kd = &k_s[r_ld * LDT + c_ld];
            unsigned short* vd = &v_s[r_ld * LDT + c_ld];
#pragma unroll
            for (int j = 0; j < 4; ++j) {
                copy16B(ks + j * 8, kd + j * 8);
                copy16B(vs + j * 8, vd + j * 8);
            }
        }
        ssq[tid] = 0.0f;
        wait_async_lds();
        __syncthreads();

        // ---- scores s[h][q][k] = (q_h . k_h) / sqrt(D): 2 chained WMMAs
        v8f c8 = {};
#pragma unroll
        for (int s = 0; s < 2; ++s) {
            union { v16bf v; unsigned int u[8]; } bfrag;
            int col = lane & 15;
            int kb2 = (lane < 16) ? 0 : 16;
            const unsigned int* p =
                (const unsigned int*)&k_s[col * LDT + h * 64 + s * 32 + kb2];
#pragma unroll
            for (int i = 0; i < 8; ++i) bfrag.u[i] = p[i];
            c8 = __builtin_amdgcn_wmma_f32_16x16x32_bf16(
                false, (s == 0) ? a0.v : a1.v, false, bfrag.v,
                (short)0, c8, false, false);
        }

        // ---- cross-head RMS reduction via LDS float atomics
        float sc[8];
        const int qq = (lane < 16) ? 0 : 8;
        const int kc = lane & 15;
#pragma unroll
        for (int r = 0; r < 8; ++r) {
            sc[r] = c8[r] * 0.125f;                       // 1/sqrt(64)
            atomicAdd(&ssq[(qq + r) * 16 + kc], sc[r] * sc[r]);
        }
        __syncthreads();
#pragma unroll
        for (int r = 0; r < 8; ++r) {
            float w = sc[r] * rsqrtf(ssq[(qq + r) * 16 + kc] * 0.125f + 1e-8f);
            w_s[h * 256 + (qq + r) * 16 + kc] = w;
        }
        __syncthreads();

        // ---- hypernet mix: x[q,k,d] = sum_h w*v; silu; out += w*x
#pragma unroll 1
        for (int k2 = 0; k2 < 16; ++k2) {
            float wv[8];
#pragma unroll
            for (int hh = 0; hh < 8; ++hh) wv[hh] = w_s[hh * 256 + q_my * 16 + k2];
            float x[4] = {0.f, 0.f, 0.f, 0.f};
#pragma unroll
            for (int hh = 0; hh < 8; ++hh) {
                const unsigned short* vp = &v_s[k2 * LDT + hh * 64 + d_my];
#pragma unroll
                for (int j = 0; j < 4; ++j) x[j] += wv[hh] * bf16_to_f32(vp[j]);
            }
#pragma unroll
            for (int j = 0; j < 4; ++j) {
                float xv = x[j];
                x[j] = xv / (1.0f + __expf(-xv));          // silu
            }
#pragma unroll
            for (int hh = 0; hh < 8; ++hh)
#pragma unroll
                for (int j = 0; j < 4; ++j) oacc[hh][j] += wv[hh] * x[j];
        }
        __syncthreads();
    }

    // ---- write attn output (bf16) for final projection
    unsigned short* orow = Obf + qrow0 + (size_t)q_my * E;
#pragma unroll
    for (int hh = 0; hh < 8; ++hh)
#pragma unroll
        for (int j = 0; j < 4; ++j)
            orow[hh * 64 + d_my + j] = f32_to_bf16(oacc[hh][j]);
}

// ---------------------------------------------------------------------------
extern "C" void kernel_launch(void* const* d_in, const int* in_sizes, int n_in,
                              void* d_out, int out_size, void* d_ws, size_t ws_size,
                              hipStream_t stream) {
    const float* query = (const float*)d_in[0];
    const float* key   = (const float*)d_in[1];
    const float* value = (const float*)d_in[2];
    const float* Wq    = (const float*)d_in[3];
    const float* bq    = (const float*)d_in[4];
    const float* Wk    = (const float*)d_in[5];
    const float* bk    = (const float*)d_in[6];
    const float* Wv    = (const float*)d_in[7];
    const float* bv    = (const float*)d_in[8];
    const float* Wo    = (const float*)d_in[9];
    const float* bo    = (const float*)d_in[10];

    const int M = 2048, E = 512;            // B*L rows, embed dim
    unsigned short* qb  = (unsigned short*)d_ws;
    unsigned short* kb  = qb + (size_t)M * E;
    unsigned short* vb  = kb + (size_t)M * E;
    unsigned short* ab  = vb + (size_t)M * E;     // attn output (bf16)
    unsigned short* wqb = ab + (size_t)M * E;     // bf16 weights
    unsigned short* wkb = wqb + (size_t)E * E;
    unsigned short* wvb = wkb + (size_t)E * E;
    unsigned short* wob = wvb + (size_t)E * E;

    // one-shot weight conversion (E*E = 262144 elems each, 4/thread)
    dim3 cg(E * E / 1024), blk(256);
    cvt_f32_bf16_kernel<<<cg, blk, 0, stream>>>(Wq, wqb, E * E);
    cvt_f32_bf16_kernel<<<cg, blk, 0, stream>>>(Wk, wkb, E * E);
    cvt_f32_bf16_kernel<<<cg, blk, 0, stream>>>(Wv, wvb, E * E);
    cvt_f32_bf16_kernel<<<cg, blk, 0, stream>>>(Wo, wob, E * E);

    dim3 gg(M / 64, E / 128);
    gemm_bt_kernel<float, true><<<gg, blk, 0, stream>>>(query, wqb, bq, qb, M, E, E);
    gemm_bt_kernel<float, true><<<gg, blk, 0, stream>>>(key,   wkb, bk, kb, M, E, E);
    gemm_bt_kernel<float, true><<<gg, blk, 0, stream>>>(value, wvb, bv, vb, M, E, E);

    attn_kernel<<<dim3(128), blk, 0, stream>>>(qb, kb, vb, ab);

    gemm_bt_kernel<unsigned short, false><<<gg, blk, 0, stream>>>(
        ab, wob, bo, (float*)d_out, M, E, E);
}
